// GraphAttentionLayer_5884105195639
// MI455X (gfx1250) — compile-verified, matrix operand order checked by reading, not run on hardware
//
#include <hip/hip_runtime.h>
#include <cstdint>

#define B_ 4
#define S_ 2048
#define F_ 128
#define H_ 4
#define D_ 128
#define HD_ 512
#define OUT_ 128
#define NEG_SLOPE 0.01f

typedef __attribute__((ext_vector_type(16))) __bf16 v16bf;
typedef __attribute__((ext_vector_type(8)))  float  v8f;

union Frag { v16bf v; unsigned u[8]; };

// ---- async global->LDS (CDNA5, ASYNCcnt) with compile-time fallback --------
#if defined(__gfx1250__) && __has_builtin(__builtin_amdgcn_global_load_async_to_lds_b128) && __has_builtin(__builtin_amdgcn_s_wait_asynccnt)
#define TRY_ASYNC 1
#else
#define TRY_ASYNC 0
#endif

#if TRY_ASYNC
typedef int v4i_ __attribute__((vector_size(4 * sizeof(int))));
__device__ __forceinline__ void async_b128(void* g, void* l) {
  __builtin_amdgcn_global_load_async_to_lds_b128((v4i_*)g, (v4i_*)l, 0, 0);
}
__device__ __forceinline__ void async_wait() { __builtin_amdgcn_s_wait_asynccnt(0); }
#else
__device__ __forceinline__ void async_b128(void* g, void* l) {
  *(uint4*)l = *(const uint4*)g;
}
__device__ __forceinline__ void async_wait() {}
#endif

__device__ __forceinline__ unsigned short f2bf(float x) {
  unsigned u = __float_as_uint(x);
  unsigned r = u + 0x7FFFu + ((u >> 16) & 1u);   // round-to-nearest-even
  return (unsigned short)(r >> 16);
}
__device__ __forceinline__ float bf2f(unsigned short h) {
  return __uint_as_float(((unsigned)h) << 16);
}
__device__ __forceinline__ float wave_max(float x) {
#pragma unroll
  for (int o = 16; o > 0; o >>= 1) x = fmaxf(x, __shfl_xor(x, o, 32));
  return x;
}
__device__ __forceinline__ float wave_sum(float x) {
#pragma unroll
  for (int o = 16; o > 0; o >>= 1) x += __shfl_xor(x, o, 32);
  return x;
}
// dword index (in a 16-dword = 32-bf16 row) of the (k,k+1) pair for fragment
// VGPR v on this lane (ISA 7.12.2 16-bit 16x32 layout).
__device__ __forceinline__ int kdw(int v, int lane) {
  return (v & 3) + ((v >> 2) << 3) + (((lane >> 4) & 1) << 2);
}
__device__ __forceinline__ float lrelu(float x) { return x > 0.f ? x : x * NEG_SLOPE; }

// ---------------------------------------------------------------- K0: f32->bf16 weights
__global__ void convert_kernel(const float* __restrict__ W_w, const float* __restrict__ out_w,
                               unsigned short* __restrict__ Wbf, unsigned short* __restrict__ OWbf) {
  int i = blockIdx.x * blockDim.x + threadIdx.x;
  if (i < HD_ * F_) {
    Wbf[i] = f2bf(W_w[i]);
  } else {
    int j = i - HD_ * F_;
    if (j < OUT_ * HD_) OWbf[j] = f2bf(out_w[j]);
  }
}

// ---------------------------------------------------------------- K1/K4: WMMA GEMM, 16x64 tile per wave
// C[m,n] = act( sum_k A[m,k]*Wbf[n,k] + bias[n] )
// MODE 0: bf16 stores to hp (B,H,S,D) and hpT (B,H,D,S).  MODE 1: fp32 relu, row-major MxN.
template <int MODE>
__global__ __launch_bounds__(32)
void gemm_wmma_kernel(const float* __restrict__ A, const unsigned short* __restrict__ Wbf,
                      const float* __restrict__ bias, void* __restrict__ C0,
                      void* __restrict__ C1, int M, int N, int K) {
  __shared__ unsigned short aT[16 * 32];   // A tile  (bf16, converted)
  __shared__ unsigned short wT[64 * 32];   // W tile: 64 rows(n) x 32 k
  const int lane = threadIdx.x;
  const int n0 = blockIdx.x * 64;
  const int m0 = blockIdx.y * 16;
  unsigned* aT32 = (unsigned*)aT;
  unsigned* wT32 = (unsigned*)wT;
  v8f acc[4] = {};

  for (int k0 = 0; k0 < K; k0 += 32) {
    // stage A tile 16x32 (f32 -> bf16), coalesced
#pragma unroll
    for (int r = 0; r < 16; ++r)
      aT[r * 32 + lane] = f2bf(A[(size_t)(m0 + r) * K + k0 + lane]);
    // stage W tile 64x32 bf16: 256 x b128 chunks, 8 per lane (async -> LDS)
#pragma unroll
    for (int t = 0; t < 8; ++t) {
      int idx = t * 32 + lane;
      int r = idx >> 2, c = idx & 3;
      async_b128((void*)(Wbf + (size_t)(n0 + r) * K + k0 + c * 8),
                 (char*)wT + r * 64 + c * 16);
    }
    if (k0 + 32 < K)
      __builtin_prefetch(&A[(size_t)(m0 + (lane & 15)) * K + k0 + 32], 0, 0);
    async_wait();
    __syncthreads();

    Frag af;
#pragma unroll
    for (int v = 0; v < 8; ++v) af.u[v] = aT32[(lane & 15) * 16 + kdw(v, lane)];
#pragma unroll
    for (int nt = 0; nt < 4; ++nt) {
      Frag bf;
#pragma unroll
      for (int v = 0; v < 8; ++v)
        bf.u[v] = wT32[(nt * 16 + (lane & 15)) * 16 + kdw(v, lane)];
      acc[nt] = __builtin_amdgcn_wmma_f32_16x16x32_bf16(false, af.v, false, bf.v,
                                                        (short)0, acc[nt], false, false);
    }
    __syncthreads();
  }

  const int nn = lane & 15;
#pragma unroll
  for (int nt = 0; nt < 4; ++nt) {
    const int n = n0 + nt * 16 + nn;
    const float bn = bias[n];
#pragma unroll
    for (int v = 0; v < 8; ++v) {
      int mrow = m0 + v + ((lane >> 4) & 1) * 8;   // C layout: VGPRv -> M = v (+8 upper half)
      float val = acc[nt][v] + bn;
      if (MODE == 0) {
        int b = mrow >> 11, s = mrow & (S_ - 1);
        int hh = n >> 7, d = n & (D_ - 1);
        unsigned short bv = f2bf(val);
        ((unsigned short*)C0)[(((size_t)b * H_ + hh) * S_ + s) * D_ + d] = bv;   // hp  (B,H,S,D)
        ((unsigned short*)C1)[(((size_t)b * H_ + hh) * D_ + d) * S_ + s] = bv;   // hpT (B,H,D,S)
      } else {
        ((float*)C0)[(size_t)mrow * N + n] = fmaxf(val, 0.f);
      }
    }
  }
}

// ---------------------------------------------------------------- K2: e1/e2
__global__ __launch_bounds__(32)
void e_kernel(const unsigned short* __restrict__ hp, const float* __restrict__ a_w,
              float* __restrict__ e1, float* __restrict__ e2) {
  const int row = blockIdx.x;       // (b*H + h)*S + s
  const int lane = threadIdx.x;
  const unsigned* hp32 = (const unsigned*)hp;
  unsigned w0 = hp32[(size_t)row * 64 + lane * 2];
  unsigned w1 = hp32[(size_t)row * 64 + lane * 2 + 1];
  float x0 = bf2f((unsigned short)w0), x1 = bf2f((unsigned short)(w0 >> 16));
  float x2 = bf2f((unsigned short)w1), x3 = bf2f((unsigned short)(w1 >> 16));
  int d0 = lane * 4;
  float s1 = x0 * a_w[d0] + x1 * a_w[d0 + 1] + x2 * a_w[d0 + 2] + x3 * a_w[d0 + 3];
  float s2 = x0 * a_w[D_ + d0] + x1 * a_w[D_ + d0 + 1] + x2 * a_w[D_ + d0 + 2] + x3 * a_w[D_ + d0 + 3];
  s1 = wave_sum(s1);
  s2 = wave_sum(s2);
  if (lane == 0) { e1[row] = s1; e2[row] = s2; }
}

// ---------------------------------------------------------------- K3: fused softmax + attn write + attn@hp
__global__ __launch_bounds__(32)
void attn_kernel(const unsigned short* __restrict__ hpT, const float* __restrict__ e1g,
                 const float* __restrict__ e2g, const unsigned char* __restrict__ mask,
                 const float* __restrict__ a_b, float* __restrict__ attn,
                 float* __restrict__ ctx) {
  __shared__ float fRow[S_];                 // e2[j]+b, -inf where masked   (8 KB)
  __shared__ unsigned short aT[16 * 32];     // attn tile bf16 (A fragments) (1 KB)
  __shared__ unsigned short hpL[D_ * 32];    // hpT tile [d][k]              (8 KB)
  const int lane = threadIdx.x;
  const int it = blockIdx.x, h = blockIdx.y, b = blockIdx.z;
  const int bh = b * H_ + h;
  const int i0 = it * 16;
  const float ab = a_b[0];
  const float NEGINF = -__builtin_inff();

  for (int t = 0; t < S_ / 32; ++t) {
    int j = t * 32 + lane;
    fRow[j] = mask[b * S_ + j] ? NEGINF : (e2g[(size_t)bh * S_ + j] + ab);
  }
  __syncthreads();

  float e1v[16], rowM[16], rowInv[16];
#pragma unroll
  for (int r = 0; r < 16; ++r) {
    e1v[r] = e1g[(size_t)bh * S_ + i0 + r];
    float m = -3.0e38f;
    for (int t = 0; t < S_ / 32; ++t)
      m = fmaxf(m, lrelu(e1v[r] + fRow[t * 32 + lane]));
    m = wave_max(m);
    float s = 0.f;
    for (int t = 0; t < S_ / 32; ++t)
      s += __expf(lrelu(e1v[r] + fRow[t * 32 + lane]) - m);
    s = wave_sum(s);
    rowM[r] = m;
    rowInv[r] = 1.f / s;
  }

  unsigned* aT32 = (unsigned*)aT;
  unsigned* hpL32 = (unsigned*)hpL;
  v8f acc[8] = {};

  for (int j0 = 0; j0 < S_; j0 += 32) {
    // attn values for 16x32 tile: coalesced fp32 store + bf16 A staging
#pragma unroll
    for (int r = 0; r < 16; ++r) {
      int j = j0 + lane;
      float sc = lrelu(e1v[r] + fRow[j]);
      float a = __expf(sc - rowM[r]) * rowInv[r];     // masked -> exp(-inf)=0
      attn[((size_t)bh * S_ + i0 + r) * S_ + j] = a;
      aT[r * 32 + lane] = f2bf(a);
    }
    // stage hpT rows d=0..127, k = j0..j0+31: linear async copy, 512 x b128
#pragma unroll
    for (int t = 0; t < 16; ++t) {
      int idx = t * 32 + lane;
      int d = idx >> 2, c = idx & 3;
      async_b128((void*)(hpT + ((size_t)bh * D_ + d) * S_ + j0 + c * 8),
                 (char*)hpL + d * 64 + c * 16);
    }
    async_wait();
    __syncthreads();

    Frag af;
#pragma unroll
    for (int v = 0; v < 8; ++v) af.u[v] = aT32[(lane & 15) * 16 + kdw(v, lane)];
#pragma unroll
    for (int dt = 0; dt < 8; ++dt) {
      Frag bf;
#pragma unroll
      for (int v = 0; v < 8; ++v)
        bf.u[v] = hpL32[(dt * 16 + (lane & 15)) * 16 + kdw(v, lane)];
      acc[dt] = __builtin_amdgcn_wmma_f32_16x16x32_bf16(false, af.v, false, bf.v,
                                                        (short)0, acc[dt], false, false);
    }
    __syncthreads();
  }

  // ctx tile 16x128 -> (B,S,H*D)  (0,2,1,3 transpose via indexing)
#pragma unroll
  for (int dt = 0; dt < 8; ++dt) {
#pragma unroll
    for (int v = 0; v < 8; ++v) {
      int row = i0 + v + ((lane >> 4) & 1) * 8;
      int col = h * D_ + dt * 16 + (lane & 15);
      ctx[((size_t)b * S_ + row) * HD_ + col] = acc[dt][v];
    }
  }
}

// ---------------------------------------------------------------- launch
extern "C" void kernel_launch(void* const* d_in, const int* in_sizes, int n_in,
                              void* d_out, int out_size, void* d_ws, size_t ws_size,
                              hipStream_t stream) {
  const float* h            = (const float*)d_in[0];
  const unsigned char* mask = (const unsigned char*)d_in[1];   // jnp.bool_ = 1 byte
  const float* W_w          = (const float*)d_in[2];
  const float* W_b          = (const float*)d_in[3];
  const float* a_w          = (const float*)d_in[4];
  const float* a_b          = (const float*)d_in[5];
  const float* out_w        = (const float*)d_in[6];
  const float* out_b        = (const float*)d_in[7];

  char* ws = (char*)d_ws;
  unsigned short* hp_bf = (unsigned short*)ws;  ws += (size_t)B_ * H_ * S_ * D_ * 2;  //  8 MB (B,H,S,D)
  unsigned short* hpT   = (unsigned short*)ws;  ws += (size_t)B_ * H_ * D_ * S_ * 2;  //  8 MB (B,H,D,S)
  float* ctx = (float*)ws;                      ws += (size_t)B_ * S_ * HD_ * 4;      // 16 MB
  float* e1 = (float*)ws;                       ws += (size_t)B_ * H_ * S_ * 4;
  float* e2 = (float*)ws;                       ws += (size_t)B_ * H_ * S_ * 4;
  unsigned short* Wbf  = (unsigned short*)ws;   ws += (size_t)HD_ * F_ * 2;
  unsigned short* OWbf = (unsigned short*)ws;   ws += (size_t)OUT_ * HD_ * 2;

  float* outp  = (float*)d_out;                         // (B,S,OUT)
  float* attnp = outp + (size_t)B_ * S_ * OUT_;         // (B,H,S,S)

  convert_kernel<<<(HD_ * F_ + OUT_ * HD_ + 255) / 256, 256, 0, stream>>>(W_w, out_w, Wbf, OWbf);
  gemm_wmma_kernel<0><<<dim3(HD_ / 64, (B_ * S_) / 16), 32, 0, stream>>>(
      h, Wbf, W_b, hp_bf, hpT, B_ * S_, HD_, F_);
  e_kernel<<<B_ * H_ * S_, 32, 0, stream>>>(hp_bf, a_w, e1, e2);
  attn_kernel<<<dim3(S_ / 16, H_, B_), 32, 0, stream>>>(hpT, e1, e2, mask, a_b, attnp, ctx);
  gemm_wmma_kernel<1><<<dim3(OUT_ / 64, (B_ * S_) / 16), 32, 0, stream>>>(
      ctx, OWbf, out_b, d_out, nullptr, B_ * S_, OUT_, HD_);
}
